// MetaRNNBase_85237920956817
// MI455X (gfx1250) — matrix operand order repaired and checked by reading, not compiled, not run on hardware
//
#include <hip/hip_runtime.h>
#include <hip/hip_bf16.h>
#include <math.h>

// ---------------------------------------------------------------------------
// MetaRNN (HyperNetwork RNN) for MI455X / gfx1250.
//
// Design:
//  * All GEMMs via V_WMMA_F32_16X16X32_BF16 (bf16 operands, f32 accumulate).
//  * The T=512 serial scan is batch-row-local -> one persistent workgroup per
//    16-row batch tile runs the WHOLE scan (512 steps) with LDS-resident
//    carries and __syncthreads() between stages.  11 kernel launches total.
//  * Per step, per block: hyper-cell GEMM (K=640), recurrent GEMM (K=256),
//    z GEMM (K=128), then per-wave 3-gate d GEMMs (K=32) whose accumulators
//    feed the tanh epilogue directly (d never touches memory).
//  * global_prefetch of step t+1's x_t / xW_t streams issued during step t
//    to hide the only cold-miss latency on the serial critical path.
//  * Workspace: xbfA 67MB + xbfB 67MB + xW 134MB + ~1MB weights  (~270MB).
// ---------------------------------------------------------------------------

typedef __bf16 bf16;
typedef __attribute__((ext_vector_type(16))) __bf16 v16bf;
typedef __attribute__((ext_vector_type(8)))  __bf16 v8bf;
typedef __attribute__((ext_vector_type(8)))  float  v8f;

constexpr int kB  = 256;   // batch
constexpr int kT  = 512;   // time
constexpr int kD  = 256;   // input dim
constexpr int kH  = 256;   // main hidden
constexpr int kHh = 128;   // hyper hidden
constexpr int kE  = 32;    // embedding
constexpr int kL  = 2;     // layers

#define DEV static __device__ __forceinline__

// ---- WMMA fragment helpers (contiguous-k A/B order, wave32) --------------

// A fragment: row-major [16, K] tile; lane&15 = M row, lane>>4 picks K range
// {0..15} vs {16..31}.  Works for global or LDS pointers.
DEV v16bf load_a_frag(const bf16* A, int lda, int k0) {
  const int lane = threadIdx.x & 31;
  const int r  = lane & 15;
  const int kh = (lane >> 4) << 4;               // 0 or 16
  const bf16* p = A + (size_t)r * lda + k0 + kh;
  return *(const v16bf*)(p);                     // K = k0+kh .. +15
}

// B fragment for C = A * W^T with W row-major [N, K]:
// lane&15 = N column (row of W); lane>>4 selects K range {0..15} vs {16..31}.
DEV v16bf load_b_frag(const bf16* W, int ldw, int n0, int k0) {
  const int lane = threadIdx.x & 31;
  const int c  = lane & 15;
  const int kh = (lane >> 4) << 4;               // 0 or 16
  return *(const v16bf*)(W + (size_t)(n0 + c) * ldw + k0 + kh);
}

DEV v8f wmma_bf16(v16bf a, v16bf b, v8f c) {
  return __builtin_amdgcn_wmma_f32_16x16x32_bf16(
      /*neg_a=*/false, a, /*neg_b=*/false, b,
      /*c_mod=*/(short)0, c, /*reuse_a=*/false, /*reuse_b=*/false);
}

// C/D layout: element r <-> (M = r + 8*(lane>>4), N = lane&15).
DEV void store_c_f32(float* C, int ldc, v8f c) {
  const int lane = threadIdx.x & 31;
  const int col = lane & 15;
  const int rb  = (lane >> 4) << 3;
#pragma unroll
  for (int r = 0; r < 8; ++r) C[(size_t)(rb + r) * ldc + col] = c[r];
}

DEV void store_c_tanh_bf16(bf16* C, int ldc, v8f c, const float* bias) {
  const int lane = threadIdx.x & 31;
  const int col = lane & 15;
  const int rb  = (lane >> 4) << 3;
  const float bv = bias[col];
#pragma unroll
  for (int r = 0; r < 8; ++r)
    C[(size_t)(rb + r) * ldc + col] = (bf16)tanhf(c[r] + bv);
}

// ---- Utility ---------------------------------------------------------------

__global__ void k_cvt_bf16(const float* __restrict__ src, bf16* __restrict__ dst, int n) {
  int i = blockIdx.x * blockDim.x + threadIdx.x;
  if (i < n) dst[i] = (bf16)src[i];
}

// ---- Hoisted input GEMM: xW = x . Wih^T  (M=B*T, N=H=256, K=256) -----------
// Full-device kernel: (B*T)/16 row tiles, 4 waves/block, 64 cols per wave.
__global__ void k_xw(const bf16* __restrict__ xin,   // [B*T, 256]
                     const bf16* __restrict__ Wih,   // [H, 256]
                     float* __restrict__ xW)         // [B*T, H]
{
  const int wave = threadIdx.x >> 5;                 // 0..3
  const size_t mrow = (size_t)blockIdx.x * 16;
  const bf16* A = xin + mrow * kD;
  const int n0 = wave * 64;
  v8f c[4] = {};
  for (int k = 0; k < kD; k += 32) {
    v16bf a = load_a_frag(A, kD, k);
#pragma unroll
    for (int j = 0; j < 4; ++j)
      c[j] = wmma_bf16(a, load_b_frag(Wih, kD, n0 + 16 * j, k), c[j]);
  }
#pragma unroll
  for (int j = 0; j < 4; ++j)
    store_c_f32(xW + mrow * kH + n0 + 16 * j, kH, c[j]);
}

// ---- Persistent scan kernel ------------------------------------------------
// grid = 16 blocks (one per 16-row batch tile), 256 threads (8 waves).
// Carries h (bf16) and hh (bf16, ping-pong) live in LDS for all 512 steps.
__global__ void k_scan(const bf16*  __restrict__ xin,     // [B, T, 256] layer input
                       const float* __restrict__ xW,      // [B, T, H] hoisted GEMM
                       const bf16*  __restrict__ Wih_hy,  // [HH, D+H]
                       const bf16*  __restrict__ Whh_hy,  // [HH, HH]
                       const float* __restrict__ b_hy,    // [HH]
                       const bf16*  __restrict__ Whh,     // [H, H]
                       const bf16*  __restrict__ Whz,     // [3, E, HH]
                       const float* __restrict__ bz,      // [3, E]
                       const bf16*  __restrict__ Wzd,     // [3, H, E]
                       const float* __restrict__ bias,    // [H]
                       float* __restrict__ yout,          // [B,T,H] f32 (layer 1) or null
                       bf16*  __restrict__ ybf,           // [B,T,H] bf16 (layer 0) or null
                       float* __restrict__ hlast)         // [B,H] final hidden or null
{
  __shared__ bf16  h_lds[16 * 256];                 // main hidden carry
  __shared__ bf16  hh_lds[2][16 * 128];             // hyper hidden carry (ping-pong)
  __shared__ bf16  zs[16 * 96];                     // z embeddings (bf16 A-fragments)
  __shared__ float u_lds[16 * 256];                 // u = h . Whh^T

  const int mtile = blockIdx.x;                     // 0..15
  const int tid   = threadIdx.x;
  const int wave  = tid >> 5;                       // 0..7
  const int lane  = tid & 31;
  const int ldx   = kT * kD;                        // batch stride of x

  // zero-init carries
  for (int i = tid; i < 16 * 256; i += 256) h_lds[i] = (bf16)0.0f;
  for (int i = tid; i < 16 * 128; i += 256) {
    hh_lds[0][i] = (bf16)0.0f;
    hh_lds[1][i] = (bf16)0.0f;
  }
  __syncthreads();

  const bf16* Axg = xin + (size_t)(mtile * 16) * ldx;

#pragma unroll 1
  for (int t = 0; t < kT; ++t) {
    const bf16* hh_cur = hh_lds[t & 1];
    bf16*       hh_nxt = hh_lds[(t + 1) & 1];

    // ---- prefetch next step's cold streams (fire-and-forget) --------------
    // x_{t+1}: 16 rows x 512B (4 x 128B lines/row)  -> threads 0..63
    // xW_{t+1}: 16 rows x 1024B (8 x 128B lines/row) -> threads 64..191
    if (t + 1 < kT) {
      if (tid < 64) {
        const bf16* px = Axg + (size_t)(tid >> 2) * ldx
                       + (size_t)(t + 1) * kD + (tid & 3) * 64;
        __builtin_prefetch(px, 0, 1);
      } else if (tid < 192) {
        const int q = tid - 64;
        const float* pw = xW + ((size_t)(mtile * 16 + (q >> 3)) * kT + (t + 1)) * kH
                        + (q & 7) * 32;
        __builtin_prefetch(pw, 0, 1);
      }
    }

    // ---- stage A: hyper cell (waves 0-3) + recurrent u (waves 4-7) --------
    if (wave < 4) {
      // hh_new = tanh([x_t, h] Wih_hy^T + hh Whh_hy^T + b_hy), 32 cols/wave
      const int n0 = wave * 32;
      const bf16* Ax = Axg + (size_t)t * kD;
      v8f c[2] = {};
      for (int k = 0; k < kD; k += 32) {            // x_t part (K = D)
        v16bf a = load_a_frag(Ax, ldx, k);
#pragma unroll
        for (int j = 0; j < 2; ++j)
          c[j] = wmma_bf16(a, load_b_frag(Wih_hy, kD + kH, n0 + 16 * j, k), c[j]);
      }
      for (int k = 0; k < kH; k += 32) {            // h part (K = H), from LDS
        v16bf a = load_a_frag(h_lds, kH, k);
#pragma unroll
        for (int j = 0; j < 2; ++j)
          c[j] = wmma_bf16(a, load_b_frag(Wih_hy + kD, kD + kH, n0 + 16 * j, k), c[j]);
      }
      for (int k = 0; k < kHh; k += 32) {           // hyper recurrence (K = HH)
        v16bf a = load_a_frag(hh_cur, kHh, k);
#pragma unroll
        for (int j = 0; j < 2; ++j)
          c[j] = wmma_bf16(a, load_b_frag(Whh_hy, kHh, n0 + 16 * j, k), c[j]);
      }
#pragma unroll
      for (int j = 0; j < 2; ++j)
        store_c_tanh_bf16(hh_nxt + n0 + 16 * j, kHh, c[j], b_hy + n0 + 16 * j);
    } else {
      // u = h . Whh^T, 64 cols/wave, A from LDS
      const int n0 = (wave - 4) * 64;
      v8f c[4] = {};
      for (int k = 0; k < kH; k += 32) {
        v16bf a = load_a_frag(h_lds, kH, k);
#pragma unroll
        for (int j = 0; j < 4; ++j)
          c[j] = wmma_bf16(a, load_b_frag(Whh, kH, n0 + 16 * j, k), c[j]);
      }
#pragma unroll
      for (int j = 0; j < 4; ++j)
        store_c_f32(u_lds + n0 + 16 * j, kH, c[j]);
    }
    __syncthreads();

    // ---- stage B1: z = hh_new . Whz^T + bz  (6 waves, one 16x16 tile each)
    if (wave < 6) {
      const int g   = wave >> 1;
      const int ntz = wave & 1;
      v8f c = {};
      for (int k = 0; k < kHh; k += 32) {
        v16bf a = load_a_frag(hh_nxt, kHh, k);
        c = wmma_bf16(a, load_b_frag(Whz + (size_t)g * kE * kHh, kHh, ntz * 16, k), c);
      }
      const int col = lane & 15;
      const int rb  = (lane >> 4) << 3;
      const float bv = bz[g * kE + ntz * 16 + col];
#pragma unroll
      for (int r = 0; r < 8; ++r)
        zs[(rb + r) * 96 + g * 32 + ntz * 16 + col] = (bf16)(c[r] + bv);
    }
    __syncthreads();

    // ---- stage B2+C: d_g = z_g . Wzd_g^T (K=32) for all 3 gates kept in
    // accumulators of the SAME wave, then tanh epilogue directly on them.
    {
      v16bf a0 = load_a_frag(zs + 0,  96, 0);
      v16bf a1 = load_a_frag(zs + 32, 96, 0);
      v16bf a2 = load_a_frag(zs + 64, 96, 0);
      const int col = lane & 15;
      const int rb  = (lane >> 4) << 3;
#pragma unroll 1
      for (int i = 0; i < 2; ++i) {
        const int nt = wave * 2 + i;                // 16 col tiles over 8 waves
        v8f c0 = {}, c1 = {}, c2 = {};
        c0 = wmma_bf16(a0, load_b_frag(Wzd + 0 * (size_t)kH * kE, kE, nt * 16, 0), c0);
        c1 = wmma_bf16(a1, load_b_frag(Wzd + 1 * (size_t)kH * kE, kE, nt * 16, 0), c1);
        c2 = wmma_bf16(a2, load_b_frag(Wzd + 2 * (size_t)kH * kE, kE, nt * 16, 0), c2);
        const int j = nt * 16 + col;
        const float bj = bias[j];
#pragma unroll
        for (int r = 0; r < 8; ++r) {
          const int row = rb + r;
          const int b   = mtile * 16 + row;
          const size_t o = ((size_t)b * kT + t) * kH + j;
          const float v = tanhf(c0[r] * xW[o] + c1[r] * u_lds[row * kH + j]
                                + c2[r] + bj);
          h_lds[row * kH + j] = (bf16)v;
          if (ybf)   ybf[o]  = (bf16)v;
          if (yout)  yout[o] = v;
          if (hlast && t == kT - 1) hlast[(size_t)b * kH + j] = v;
        }
      }
    }
    __syncthreads();   // h_lds / zs / u_lds consumed before next step rewrites
  }
}

// ---------------------------------------------------------------------------
extern "C" void kernel_launch(void* const* d_in, const int* in_sizes, int n_in,
                              void* d_out, int out_size, void* d_ws, size_t ws_size,
                              hipStream_t stream) {
  (void)in_sizes; (void)n_in; (void)out_size; (void)ws_size;

  const float* x      = (const float*)d_in[0];  // [B,T,D]
  const float* Wih    = (const float*)d_in[1];  // [L,H,D]
  const float* Whh    = (const float*)d_in[2];  // [L,H,H]
  const float* bmain  = (const float*)d_in[3];  // [L,H]
  const float* Wih_hy = (const float*)d_in[4];  // [L,HH,D+H]
  const float* Whh_hy = (const float*)d_in[5];  // [L,HH,HH]
  const float* b_hy   = (const float*)d_in[6];  // [L,HH]
  const float* Whz    = (const float*)d_in[7];  // [L,3,E,HH]
  const float* bz     = (const float*)d_in[8];  // [L,3,E]
  const float* Wzd    = (const float*)d_in[9];  // [L,3,H,E]

  float* out_seq  = (float*)d_out;                       // [B,T,H]
  float* out_last = out_seq + (size_t)kB * kT * kH;      // [B,H]

  // ---- workspace carve-up --------------------------------------------------
  char* p = (char*)d_ws;
  auto alloc = [&](size_t bytes) -> void* {
    void* r = (void*)p;
    p += (bytes + 255) & ~(size_t)255;
    return r;
  };
  const size_t BT = (size_t)kB * kT;
  bf16* xbfA      = (bf16*)alloc(BT * kD * 2);           // layer-0 input, bf16
  bf16* xbfB      = (bf16*)alloc(BT * kH * 2);           // layer-0 output / layer-1 input
  float* xW       = (float*)alloc(BT * kH * 4);          // hoisted input GEMM
  bf16* Wih_bf    = (bf16*)alloc((size_t)kL * kH * kD * 2);
  bf16* Whh_bf    = (bf16*)alloc((size_t)kL * kH * kH * 2);
  bf16* Wih_hy_bf = (bf16*)alloc((size_t)kL * kHh * (kD + kH) * 2);
  bf16* Whh_hy_bf = (bf16*)alloc((size_t)kL * kHh * kHh * 2);
  bf16* Whz_bf    = (bf16*)alloc((size_t)kL * 3 * kE * kHh * 2);
  bf16* Wzd_bf    = (bf16*)alloc((size_t)kL * 3 * kH * kE * 2);

  // ---- one-time f32 -> bf16 conversions -----------------------------------
  auto cvt = [&](const float* s, bf16* d, size_t n) {
    k_cvt_bf16<<<dim3((unsigned)((n + 255) / 256)), dim3(256), 0, stream>>>(s, d, (int)n);
  };
  cvt(x,      xbfA,      BT * kD);
  cvt(Wih,    Wih_bf,    (size_t)kL * kH * kD);
  cvt(Whh,    Whh_bf,    (size_t)kL * kH * kH);
  cvt(Wih_hy, Wih_hy_bf, (size_t)kL * kHh * (kD + kH));
  cvt(Whh_hy, Whh_hy_bf, (size_t)kL * kHh * kHh);
  cvt(Whz,    Whz_bf,    (size_t)kL * 3 * kE * kHh);
  cvt(Wzd,    Wzd_bf,    (size_t)kL * 3 * kH * kE);

  // ---- layer loop: 2 launches per layer -----------------------------------
  for (int l = 0; l < kL; ++l) {
    const bf16* xin = l ? xbfB : xbfA;                   // kD == kH == 256
    const bf16*  Wih_l    = Wih_bf    + (size_t)l * kH * kD;
    const bf16*  Whh_l    = Whh_bf    + (size_t)l * kH * kH;
    const float* b_l      = bmain     + (size_t)l * kH;
    const bf16*  Wih_hy_l = Wih_hy_bf + (size_t)l * kHh * (kD + kH);
    const bf16*  Whh_hy_l = Whh_hy_bf + (size_t)l * kHh * kHh;
    const float* b_hy_l   = b_hy      + (size_t)l * kHh;
    const bf16*  Whz_l    = Whz_bf    + (size_t)l * 3 * kE * kHh;
    const float* bz_l     = bz        + (size_t)l * 3 * kE;
    const bf16*  Wzd_l    = Wzd_bf    + (size_t)l * 3 * kH * kE;

    // hoisted xW GEMM for the whole sequence (full-device)
    k_xw<<<dim3((unsigned)(BT / 16)), dim3(128), 0, stream>>>(xin, Wih_l, xW);

    // persistent scan: one workgroup per 16-row batch tile
    float* yout = l ? out_seq  : nullptr;
    bf16*  ybf  = l ? nullptr  : xbfB;
    float* hl   = (l == kL - 1) ? out_last : nullptr;
    k_scan<<<dim3(16), dim3(256), 0, stream>>>(
        xin, xW, Wih_hy_l, Whh_hy_l, b_hy_l, Whh_l, Whz_l, bz_l, Wzd_l, b_l,
        yout, ybf, hl);
  }
}
